// NativeLinear4bit_28845000360310
// MI455X (gfx1250) — matrix operand order repaired and checked by reading, not compiled, not run on hardware
//
#include <hip/hip_runtime.h>

// NF4 dequant + GEMM (y = x @ W_deq^T + bias) fused, v_wmma_f32_16x16x32_f16,
// gfx1250 wave32. Double-buffered LDS (1 barrier per K-chunk), register
// prefetch pipeline, paired-f16 LUT dequant (ds_load_b32 + v_pk_mul_f16).

typedef _Float16 half2v __attribute__((ext_vector_type(2)));
typedef _Float16 half8  __attribute__((ext_vector_type(8)));
typedef _Float16 half16 __attribute__((ext_vector_type(16)));
typedef float    f32x4  __attribute__((ext_vector_type(4)));
typedef float    v8f    __attribute__((ext_vector_type(8)));
typedef int      i32x4  __attribute__((ext_vector_type(4)));

#define IN_F   4096
#define OUT_F  11008
#define ROWS   8192          // 4 * 2048
#define MT     128           // M tile per workgroup
#define NT     128           // N tile per workgroup
#define KC     64            // K chunk (== NF4 block size)
#define LDSK   72            // padded LDS row stride in halfs (144 B)

__device__ __constant__ float NF4_LUT_C[16] = {
  -1.0f, -0.6961928009986877f, -0.5250730514526367f, -0.39491719007492065f,
  -0.28444138169288635f, -0.18477343022823334f, -0.08820830285549164f, 0.0f,
  0.07107656449079514f, 0.14263366162776947f, 0.22430233657360077f,
  0.3203405737876892f, 0.4407068192958832f, 0.5905631184577942f,
  0.796090841293335f, 1.0f
};

__global__ __launch_bounds__(256)
void nf4_linear_wmma(const float* __restrict__ x,
                     const int*   __restrict__ wq,
                     const float* __restrict__ absmax,
                     const float* __restrict__ bias,
                     float*       __restrict__ out)
{
  __shared__ __align__(16) _Float16 lx[2][MT * LDSK];  // x tiles (f16), double buffered
  __shared__ __align__(16) _Float16 lw[2][NT * LDSK];  // dequant weight tiles (f16)
  __shared__ unsigned int lut2[256];                   // byte -> packed {f16 lo, f16 hi}

  const int t    = threadIdx.x;
  const int lane = t & 31;
  const int wave = t >> 5;        // 0..7
  const int wm   = wave & 3;      // 4 waves along M, 32 rows each
  const int wn   = wave >> 2;     // 2 waves along N, 64 cols each
  const int l16  = lane & 15;
  const bool lo  = lane < 16;

  const int m0 = blockIdx.x * MT;
  const int n0 = blockIdx.y * NT;

  // Paired-f16 LUT: entry c = {f16(LUT[c&15]), f16(LUT[c>>4])}; 256 threads, 1 each.
  {
    half2v p;
    p[0] = (_Float16)NF4_LUT_C[t & 15];
    p[1] = (_Float16)NF4_LUT_C[(t >> 4) & 15];
    lut2[t] = __builtin_bit_cast(unsigned int, p);
  }

  // staging: 8 threads per row, 4 passes x 32 rows
  const int sr = t >> 3;          // 0..31
  const int sc = t & 7;           // 0..7

  // software-pipeline registers (next chunk)
  f32x4 xr[4][2];
  i32x4 wr[4];
  float am[4];

  auto prefetch = [&](int k0) {
    #pragma unroll
    for (int p = 0; p < 4; ++p) {
      const int r = p * 32 + sr;
      const float* xp = x + (size_t)(m0 + r) * IN_F + k0 + sc * 8;
      xr[p][0] = ((const f32x4*)xp)[0];
      xr[p][1] = ((const f32x4*)xp)[1];
      am[p] = absmax[(size_t)(n0 + r) * (IN_F / 64) + (k0 >> 6)];
      wr[p] = *(const i32x4*)(wq + (size_t)(n0 + r) * (IN_F / 2) + (k0 >> 1) + sc * 4);
    }
  };

  v8f acc[2][4] = {};

  // One K-chunk: stage regs -> LDS[b], barrier, prefetch k0+KC, compute LDS[b].
  auto chunk = [&](int b, int k0) {
    #pragma unroll
    for (int p = 0; p < 4; ++p) {
      const int r = p * 32 + sr;
      // x: f32 -> f16
      half8 h;
      #pragma unroll
      for (int j = 0; j < 4; ++j) {
        h[j]     = (_Float16)xr[p][0][j];
        h[4 + j] = (_Float16)xr[p][1][j];
      }
      *(half8*)&lx[b][r * LDSK + sc * 8] = h;
      // w: byte code -> paired f16 * absmax (v_pk_mul_f16); codes are 0..255
      half2v amh;
      amh[0] = amh[1] = (_Float16)am[p];
      half8 g;
      #pragma unroll
      for (int j = 0; j < 4; ++j) {
        const unsigned int pr = lut2[(unsigned int)wr[p][j]];
        half2v hv = __builtin_bit_cast(half2v, pr) * amh;
        g[2 * j]     = hv[0];
        g[2 * j + 1] = hv[1];
      }
      *(half8*)&lw[b][r * LDSK + sc * 8] = g;
    }
    __syncthreads();

    // prefetch next chunk; overlaps the WMMAs below (waited at use, not here)
    if (k0 + KC < IN_F) prefetch(k0 + KC);

    #pragma unroll
    for (int ks = 0; ks < KC; ks += 32) {
      half16 A[2], B[4];
      // A fragment: 16x32 f16; lane<16: K[0..7]+K[16..23], lane>=16: K[8..15]+K[24..31]
      #pragma unroll
      for (int mi = 0; mi < 2; ++mi) {
        const int m   = wm * 32 + mi * 16 + l16;
        const int off = m * LDSK + ks + (lo ? 0 : 8);
        half8 alo = *(const half8*)&lx[b][off];
        half8 ahi = *(const half8*)&lx[b][off + 16];
        A[mi] = __builtin_shufflevector(alo, ahi,
                  0,1,2,3,4,5,6,7,8,9,10,11,12,13,14,15);
      }
      // B fragment: 32x16 f16; lane<16: K[0..7]+K[16..23], lane>=16: K[8..15]+K[24..31]
      #pragma unroll
      for (int ni = 0; ni < 4; ++ni) {
        const int n   = wn * 64 + ni * 16 + l16;
        const int off = n * LDSK + ks + (lo ? 0 : 8);
        half8 blo = *(const half8*)&lw[b][off];
        half8 bhi = *(const half8*)&lw[b][off + 16];
        B[ni] = __builtin_shufflevector(blo, bhi,
                  0,1,2,3,4,5,6,7,8,9,10,11,12,13,14,15);
      }
      #pragma unroll
      for (int mi = 0; mi < 2; ++mi)
        #pragma unroll
        for (int ni = 0; ni < 4; ++ni)
          acc[mi][ni] = __builtin_amdgcn_wmma_f32_16x16x32_f16(
              false, A[mi], false, B[ni], (short)0, acc[mi][ni], false, false);
    }
  };

  prefetch(0);
  __syncthreads();                // lut2 visible before first dequant

  for (int k0 = 0; k0 < IN_F; k0 += 2 * KC) {
    chunk(0, k0);                 // static buffer indices (loop unrolled x2)
    chunk(1, k0 + KC);
  }

  // ---- epilogue: C/D layout -> global, + bias ----
  // VGPR v: lanes 0-15 -> M = v, lanes 16-31 -> M = v + 8; N = lane % 16
  #pragma unroll
  for (int ni = 0; ni < 4; ++ni) {
    const int col   = n0 + wn * 64 + ni * 16 + l16;
    const float bcl = bias[col];
    #pragma unroll
    for (int mi = 0; mi < 2; ++mi) {
      const int mrow = m0 + wm * 32 + mi * 16 + (lo ? 0 : 8);
      #pragma unroll
      for (int v = 0; v < 8; ++v) {
        out[(size_t)(mrow + v) * OUT_F + col] = acc[mi][ni][v] + bcl;
      }
    }
  }
}

extern "C" void kernel_launch(void* const* d_in, const int* in_sizes, int n_in,
                              void* d_out, int out_size, void* d_ws, size_t ws_size,
                              hipStream_t stream) {
  const float* x      = (const float*)d_in[0];
  const int*   wq     = (const int*)d_in[1];
  const float* absmax = (const float*)d_in[2];
  const float* bias   = (const float*)d_in[3];
  float*       out    = (float*)d_out;

  dim3 grid(ROWS / MT, OUT_F / NT);   // (64, 86); M fastest -> weight tiles hot in L2
  nf4_linear_wmma<<<grid, dim3(256), 0, stream>>>(x, wq, absmax, bias, out);
}
